// AuxSelfAttention_3212635537793
// MI455X (gfx1250) — matrix operand order repaired and checked by reading, not compiled, not run on hardware
//
#include <hip/hip_runtime.h>
#include <hip/hip_bf16.h>

// Problem constants (B, T, C, H, E from the reference)
#define BB  4
#define TT  128
#define CC  256
#define HH  8
#define EE  64
#define DD  32      // C / H
#define CEE 320     // C + E

typedef __attribute__((ext_vector_type(16))) __bf16 v16bf;
typedef __attribute__((ext_vector_type(8)))  __bf16 v8bf;
typedef __attribute__((ext_vector_type(8)))  float  v8f;

// Async-copy pointer types: builtin signature is
//   (v4i addrspace(1)* src, v4i addrspace(3)* dst, imm offset, imm cpol)
typedef int v4i __attribute__((vector_size(16)));
typedef __attribute__((address_space(1))) v4i gv4i;
typedef __attribute__((address_space(3))) v4i lv4i;

// ---------------------------------------------------------------------------
// Kernel 1: per-token parts.
//   q  = x @ Wq^T + bq
//   xk = x @ Wk[:, :C]^T + bk      (the part of k that is constant over j)
//   xv = x @ Wv[:, :C]^T + bv
// ---------------------------------------------------------------------------
__global__ void __launch_bounds__(CC) prep_tokens(
    const float* __restrict__ x,
    const float* __restrict__ Wq, const float* __restrict__ bq,
    const float* __restrict__ Wk, const float* __restrict__ bk,
    const float* __restrict__ Wv, const float* __restrict__ bv,
    float* __restrict__ qG, float* __restrict__ xkG, float* __restrict__ xvG) {
  __shared__ float xr[CC];
  const int m = blockIdx.x;      // b*T + i
  const int n = threadIdx.x;     // output channel
  xr[n] = x[m * CC + n];
  __syncthreads();
  float aq = bq[n], ak = bk[n], av = bv[n];
  const float* wq = Wq + n * CC;
  const float* wk = Wk + n * CEE;
  const float* wv = Wv + n * CEE;
#pragma unroll 4
  for (int c = 0; c < CC; ++c) {
    const float xc = xr[c];
    aq = fmaf(xc, wq[c], aq);
    ak = fmaf(xc, wk[c], ak);
    av = fmaf(xc, wv[c], av);
  }
  qG[m * CC + n]  = aq;
  xkG[m * CC + n] = ak;
  xvG[m * CC + n] = av;
}

// ---------------------------------------------------------------------------
// Kernel 2: convert edge-weight slices Wk[:, C:], Wv[:, C:] to bf16, stored
// as [n][e] so a B-fragment lane reads a contiguous 16-element K-run.
// ---------------------------------------------------------------------------
__global__ void conv_weights(const float* __restrict__ Wk, const float* __restrict__ Wv,
                             __bf16* __restrict__ wkE, __bf16* __restrict__ wvE) {
  const int idx = blockIdx.x * blockDim.x + threadIdx.x;  // 0 .. CC*EE-1
  const int n = idx >> 6;
  const int e = idx & (EE - 1);
  wkE[idx] = (__bf16)Wk[n * CEE + CC + e];
  wvE[idx] = (__bf16)Wv[n * CEE + CC + e];
}

// ---------------------------------------------------------------------------
// WMMA fragment loaders (wave32, V_WMMA_F32_16X16X32_BF16).
//
// A (16x32 bf16): lane L -> row M = L&15; elements 0..15 hold
// K = 16*(L>>4) .. +15.
// Source: f32 edge tile in LDS (async-staged), converted at fragment build.
// ---------------------------------------------------------------------------
__device__ __forceinline__ v16bf load_a_frag_lds(const float* __restrict__ m,
                                                 int jbase, int kbase, int lane) {
  const int row = jbase + (lane & 15);
  const int off = (lane >> 4) * 16;
  const float* p = m + row * EE + kbase + off;
  v16bf r;
#pragma unroll
  for (int i = 0; i < 8; ++i) {
    r[i]     = (__bf16)p[i];        // K = kbase+off .. +7
    r[i + 8] = (__bf16)p[i + 8];    // K = kbase+off+8 .. +15
  }
  return r;
}

// B (32x16 bf16): lane L -> column N = nbase + (L&15); lanes 0-15 carry
// K = kbase..+15, lanes 16-31 carry K = kbase+16..+31; element index = K.
// Source: global weights stored [n][EE] bf16 -> contiguous per lane.
// ---------------------------------------------------------------------------
__device__ __forceinline__ v16bf load_b_frag(const __bf16* __restrict__ W,
                                             int nbase, int kbase, int lane) {
  const int n = nbase + (lane & 15);
  const int k = kbase + (lane >> 4) * 16;
  const __bf16* p = W + n * EE + k;
  v8bf lo = *(const v8bf*)p;
  v8bf hi = *(const v8bf*)(p + 8);
  v16bf r;
#pragma unroll
  for (int i = 0; i < 8; ++i) { r[i] = lo[i]; r[i + 8] = hi[i]; }
  return r;
}

// ---------------------------------------------------------------------------
// LDS layout for the fused kernel (dynamic shared memory, bytes)
// ---------------------------------------------------------------------------
#define SMEM_EDGE 0
#define SMEM_KB   (TT * EE * 4)                 // edge tile f32: 32 KB
#define SMEM_VB   (SMEM_KB + TT * CC * 2)       // K bf16: 64 KB
#define SMEM_Q    (SMEM_VB + TT * CC * 2)       // V bf16: 64 KB
#define SMEM_XK   (SMEM_Q + CC * 4)
#define SMEM_XV   (SMEM_XK + CC * 4)
#define SMEM_ATT  (SMEM_XV + CC * 4)
#define SMEM_Y    (SMEM_ATT + HH * TT * 4)      // logits/weights: 4 KB
#define SMEM_TOT  (SMEM_Y + CC * 4)             // total = 172032 B < 320 KB

// ---------------------------------------------------------------------------
// Kernel 3: fused per-(b,i) edge attention.
// 8 wave32s; wave w computes j-rows [16w, 16w+16) of K_i and V_i with WMMA,
// then the block does logits -> softmax -> att@V -> output projection.
// ---------------------------------------------------------------------------
__global__ void __launch_bounds__(CC) fused_edge_attn(
    const float* __restrict__ edge, const int* __restrict__ mask,
    const float* __restrict__ qG, const float* __restrict__ xkG,
    const float* __restrict__ xvG,
    const __bf16* __restrict__ wkE, const __bf16* __restrict__ wvE,
    const float* __restrict__ Wp, const float* __restrict__ bp,
    float* __restrict__ out) {
  extern __shared__ char smem[];
  float*  edgeF = (float*)(smem + SMEM_EDGE);
  __bf16* Kb    = (__bf16*)(smem + SMEM_KB);
  __bf16* Vb    = (__bf16*)(smem + SMEM_VB);
  float*  qS    = (float*)(smem + SMEM_Q);
  float*  xkS   = (float*)(smem + SMEM_XK);
  float*  xvS   = (float*)(smem + SMEM_XV);
  float*  attS  = (float*)(smem + SMEM_ATT);
  float*  yS    = (float*)(smem + SMEM_Y);

  const int bi   = blockIdx.x;        // b*T + i
  const int tid  = threadIdx.x;
  const int lane = tid & 31;
  const int wv32 = tid >> 5;          // wave id 0..7

  // Warm the L2/L0 with this thread's output-projection row (phase 6 use).
  __builtin_prefetch(Wp + (size_t)tid * CC, 0, 1);

  // ---- phase 1: stage edge[b,i] (128x64 f32) into LDS + token rows ----
  {
    const float* src = edge + (size_t)bi * (TT * EE);
#if defined(__gfx1250__) && __has_builtin(__builtin_amdgcn_global_load_async_to_lds_b128)
    // CDNA5 async copy: global -> LDS without touching VGPRs (ASYNCcnt).
    for (int it = tid; it < (TT * EE) / 4; it += CC) {
      __builtin_amdgcn_global_load_async_to_lds_b128(
          (gv4i*)(src + it * 4),
          (lv4i*)(edgeF + it * 4),
          /*offset=*/0, /*cpol=*/0);
    }
    qS[tid]  = qG[(size_t)bi * CC + tid];
    xkS[tid] = xkG[(size_t)bi * CC + tid];
    xvS[tid] = xvG[(size_t)bi * CC + tid];
#if __has_builtin(__builtin_amdgcn_s_wait_asynccnt)
    __builtin_amdgcn_s_wait_asynccnt(0);
#else
    asm volatile("s_wait_asynccnt 0x0" ::: "memory");
#endif
#else
    // Fallback: manual staged copy through VGPRs.
    const float4* src4 = (const float4*)src;
    for (int it = tid; it < (TT * EE) / 4; it += CC) {
      const float4 f = src4[it];
      float* d = edgeF + it * 4;
      d[0] = f.x; d[1] = f.y; d[2] = f.z; d[3] = f.w;
    }
    qS[tid]  = qG[(size_t)bi * CC + tid];
    xkS[tid] = xkG[(size_t)bi * CC + tid];
    xvS[tid] = xvG[(size_t)bi * CC + tid];
#endif
  }
  __syncthreads();

  // ---- phase 2: K_i / V_i = edge @ W_e^T + x-part  (WMMA bf16, f32 acc) ----
  {
    const int jb = wv32 * 16;                       // this wave's j-rows
    const v16bf a0 = load_a_frag_lds(edgeF, jb, 0,  lane);
    const v16bf a1 = load_a_frag_lds(edgeF, jb, 32, lane);
    const int nl = lane & 15;                       // D-layout: N = lane&15
    const int mh = (lane >> 4) * 8;                 // rows r / r+8 split
    for (int nt = 0; nt < 16; ++nt) {
      const int nb = nt * 16;
      // K tile
      v16bf b0 = load_b_frag(wkE, nb, 0,  lane);
      v16bf b1 = load_b_frag(wkE, nb, 32, lane);
      v8f ck = {};
      ck = __builtin_amdgcn_wmma_f32_16x16x32_bf16(false, a0, false, b0,
                                                   (short)0, ck, false, false);
      ck = __builtin_amdgcn_wmma_f32_16x16x32_bf16(false, a1, false, b1,
                                                   (short)0, ck, false, false);
      const float kxa = xkS[nb + nl];
#pragma unroll
      for (int r = 0; r < 8; ++r)
        Kb[(jb + mh + r) * CC + nb + nl] = (__bf16)(ck[r] + kxa);
      // V tile
      v16bf c0 = load_b_frag(wvE, nb, 0,  lane);
      v16bf c1 = load_b_frag(wvE, nb, 32, lane);
      v8f cv = {};
      cv = __builtin_amdgcn_wmma_f32_16x16x32_bf16(false, a0, false, c0,
                                                   (short)0, cv, false, false);
      cv = __builtin_amdgcn_wmma_f32_16x16x32_bf16(false, a1, false, c1,
                                                   (short)0, cv, false, false);
      const float vxa = xvS[nb + nl];
#pragma unroll
      for (int r = 0; r < 8; ++r)
        Vb[(jb + mh + r) * CC + nb + nl] = (__bf16)(cv[r] + vxa);
    }
  }
  __syncthreads();

  // ---- phase 3: logits att[h][j] = (q_h . k_hj) / sqrt(D), masked ----
  {
    const int* mrow = mask + (size_t)bi * TT;       // mask[b][i][:]
    for (int p = tid; p < HH * TT; p += CC) {
      const int h = p >> 7;
      const int j = p & (TT - 1);
      const __bf16* kr = Kb + j * CC + h * DD;
      const float*  qh = qS + h * DD;
      float acc = 0.f;
#pragma unroll 8
      for (int d = 0; d < DD; ++d) acc = fmaf(qh[d], (float)kr[d], acc);
      acc *= 0.17677669529663687f;                  // 1/sqrt(32)
      if (mrow[j] == 0) acc = -1e30f;
      attS[p] = acc;
    }
  }
  __syncthreads();

  // ---- phase 4: softmax; wave w owns head w (128 elems, 4 per lane) ----
  {
    float* arow = attS + wv32 * TT;
    float v0 = arow[lane], v1 = arow[lane + 32];
    float v2 = arow[lane + 64], v3 = arow[lane + 96];
    float mx = fmaxf(fmaxf(v0, v1), fmaxf(v2, v3));
#pragma unroll
    for (int off = 16; off > 0; off >>= 1) mx = fmaxf(mx, __shfl_xor(mx, off, 32));
    const bool allm = (mx <= -1e29f);               // all-masked row -> uniform
    const float e0 = allm ? 1.f : __expf(v0 - mx);
    const float e1 = allm ? 1.f : __expf(v1 - mx);
    const float e2 = allm ? 1.f : __expf(v2 - mx);
    const float e3 = allm ? 1.f : __expf(v3 - mx);
    float s = e0 + e1 + e2 + e3;
#pragma unroll
    for (int off = 16; off > 0; off >>= 1) s += __shfl_xor(s, off, 32);
    const float inv = 1.f / s;
    arow[lane]      = e0 * inv;
    arow[lane + 32] = e1 * inv;
    arow[lane + 64] = e2 * inv;
    arow[lane + 96] = e3 * inv;
  }
  __syncthreads();

  // ---- phase 5: y[c] = sum_j att[h(c)][j] * V[j][c] ----
  {
    const float* arow = attS + (tid >> 5) * TT;     // head = c / 32
    float acc = 0.f;
#pragma unroll 4
    for (int j = 0; j < TT; ++j)
      acc = fmaf(arow[j], (float)Vb[j * CC + tid], acc);
    yS[tid] = acc;
  }
  __syncthreads();

  // ---- phase 6: out = y @ Wp^T + bp ----
  {
    float acc = bp[tid];
    const float* wr = Wp + tid * CC;
#pragma unroll 4
    for (int c = 0; c < CC; ++c) acc = fmaf(yS[c], wr[c], acc);
    out[(size_t)bi * CC + tid] = acc;
  }
}

// ---------------------------------------------------------------------------
// Workspace layout (bytes): q | xk | xv | WkE(bf16) | WvE(bf16)  ~= 1.57 MB
// ---------------------------------------------------------------------------
#define WS_Q   0
#define WS_XK  (BB * TT * CC * 4)          //  524288
#define WS_XV  (WS_XK + BB * TT * CC * 4)  // 1048576
#define WS_WKE (WS_XV + BB * TT * CC * 4)  // 1572864
#define WS_WVE (WS_WKE + CC * EE * 2)      // 1605632

extern "C" void kernel_launch(void* const* d_in, const int* in_sizes, int n_in,
                              void* d_out, int out_size, void* d_ws, size_t ws_size,
                              hipStream_t stream) {
  const float* x    = (const float*)d_in[0];
  // d_in[1] = aux_x, unused by the reference
  const int*   mask = (const int*)d_in[2];
  const float* edge = (const float*)d_in[3];
  const float* Wq   = (const float*)d_in[4];
  const float* bq   = (const float*)d_in[5];
  const float* Wk   = (const float*)d_in[6];
  const float* bk   = (const float*)d_in[7];
  const float* Wv   = (const float*)d_in[8];
  const float* bv   = (const float*)d_in[9];
  const float* Wp   = (const float*)d_in[10];
  const float* bp   = (const float*)d_in[11];
  float* out = (float*)d_out;

  char* ws = (char*)d_ws;
  float*  qG  = (float*)(ws + WS_Q);
  float*  xkG = (float*)(ws + WS_XK);
  float*  xvG = (float*)(ws + WS_XV);
  __bf16* wkE = (__bf16*)(ws + WS_WKE);
  __bf16* wvE = (__bf16*)(ws + WS_WVE);

  prep_tokens<<<BB * TT, CC, 0, stream>>>(x, Wq, bq, Wk, bk, Wv, bv, qG, xkG, xvG);
  conv_weights<<<(CC * EE) / 256, 256, 0, stream>>>(Wk, Wv, wkE, wvE);
  fused_edge_attn<<<BB * TT, CC, SMEM_TOT, stream>>>(edge, mask, qG, xkG, xvG,
                                                     wkE, wvE, Wp, bp, out);
}
